// CrossDAMamba_27771258536654
// MI455X (gfx1250) — matrix operand order repaired and checked by reading, not compiled, not run on hardware
//
#include <hip/hip_runtime.h>

// ---------- WMMA types ----------
typedef __attribute__((ext_vector_type(16))) _Float16 v16h;
typedef __attribute__((ext_vector_type(8)))  float    v8f;

struct alignas(16) B128 { unsigned int w[4]; };
union Frag { v16h v; B128 b[2]; };

// ---------- problem constants ----------
// B=8, H=W=112, C=256, heads=8, d=32, ws=7, shift=3, N=49
// windows: 2048 total (win = b*256 + wy*16 + wx), rows = 2048*49 = 100352

static constexpr int NROWS   = 100352;     // 2048*49
static constexpr int NBLK    = NROWS / 16; // 6272

// workspace layout (bytes)
static constexpr size_t OFF_WQT  = 0;
static constexpr size_t OFF_WKVT = OFF_WQT  + (size_t)256*256*2;
static constexpr size_t OFF_WPT  = OFF_WKVT + (size_t)512*256*2;
static constexpr size_t OFF_BIAS = OFF_WPT  + (size_t)256*256*2;
static constexpr size_t OFF_Q    = OFF_BIAS + (size_t)8*64*64*4;
static constexpr size_t SZ_QKV   = (size_t)2048*8*64*32*2;   // 67108864
static constexpr size_t OFF_K    = OFF_Q  + SZ_QKV;
static constexpr size_t OFF_VT   = OFF_K  + SZ_QKV;
static constexpr size_t OFF_O    = OFF_VT + SZ_QKV;          // O: [2048][64][256] f16

// map a (window,token) GEMM row to the flat element offset of that row in the
// (B, H*W, C) input, applying the roll(-3,-3) + window partition gather.
__device__ __forceinline__ long long gatherRow(int row) {
    int win = row / 49, t = row - win * 49;
    int b  = win >> 8;
    int wi = win & 255;
    int wy = wi >> 4, wx = wi & 15;
    int ty = t / 7,  tx = t - ty * 7;
    int sy = wy * 7 + ty + 3; if (sy >= 112) sy -= 112;
    int sx = wx * 7 + tx + 3; if (sx >= 112) sx -= 112;
    return ((long long)b * 12544 + sy * 112 + sx) * 256;
}

// ---------- prep: transpose weights -> f16, build padded bias table ----------
__global__ __launch_bounds__(256)
void prep_kernel(const float* __restrict__ Wq, const float* __restrict__ Wkv,
                 const float* __restrict__ Wp, const float* __restrict__ btab,
                 _Float16* __restrict__ Wqt, _Float16* __restrict__ Wkvt,
                 _Float16* __restrict__ Wpt, float* __restrict__ biasArr) {
    int i = blockIdx.x * 256 + threadIdx.x;
    if (i < 65536) {                       // Wqt[n][k] = Wq[k][n]
        int n = i >> 8, k = i & 255;
        Wqt[i] = (_Float16)Wq[k * 256 + n];
        Wpt[i] = (_Float16)Wp[k * 256 + n];
    }
    if (i < 131072) {                      // Wkvt[n][k] = Wkv[k][n], n<512
        int n = i >> 8, k = i & 255;
        Wkvt[i] = (_Float16)Wkv[k * 512 + n];
    }
    if (i < 32768) {                       // bias[h][m(64)][n(64)], zero-padded
        int h = i >> 12, m = (i >> 6) & 63, n = i & 63;
        float v = 0.f;
        if (m < 49 && n < 49) {
            int ym = m / 7, xm = m - ym * 7;
            int yn = n / 7, xn = n - yn * 7;
            int rel = (ym - yn + 6) * 13 + (xm - xn + 6);
            v = btab[rel * 8 + h];
        }
        biasArr[i] = v;
    }
}

// ---------- Q projection: Q = gather(query2) @ Wq^T, *d^-0.5 ----------
__global__ __launch_bounds__(128)
void qproj_kernel(const float* __restrict__ X, const _Float16* __restrict__ Wt,
                  const float* __restrict__ bq, _Float16* __restrict__ Q) {
    const int tid = threadIdx.x, lane = tid & 31, wave = tid >> 5;
    const int rowBase = blockIdx.x * 16;
    __shared__ _Float16 As[16][32];

    const int lr = tid >> 3;             // 0..15  (A staging row)
    const int lk = (tid & 7) * 4;        // 0..28  (A staging col)
    const float* aRow = X + gatherRow(rowBase + lr) + lk;

    v8f acc[4];
    for (int i = 0; i < 4; i++) for (int j = 0; j < 8; j++) acc[i][j] = 0.f;

    const int arow = lane & 15;
    const int kbA  = (lane >> 4) * 8;    // A fragment K base
    const int kbB  = (lane >> 4) * 8;    // B fragment K base
    const int n0   = wave * 64;

    for (int kk = 0; kk < 256; kk += 32) {
        __syncthreads();
        float4 f = *(const float4*)(aRow + kk);
        if (kk + 32 < 256) __builtin_prefetch(aRow + kk + 32, 0, 3);
        As[lr][lk + 0] = (_Float16)f.x;
        As[lr][lk + 1] = (_Float16)f.y;
        As[lr][lk + 2] = (_Float16)f.z;
        As[lr][lk + 3] = (_Float16)f.w;
        __syncthreads();
        Frag A;
        A.b[0] = *(const B128*)&As[arow][kbA];
        A.b[1] = *(const B128*)&As[arow][kbA + 16];
#pragma unroll
        for (int nt = 0; nt < 4; nt++) {
            int n = n0 + nt * 16 + (lane & 15);
            const _Float16* bp = Wt + n * 256 + kk + kbB;
            Frag Bf;
            Bf.b[0] = *(const B128*)bp;
            Bf.b[1] = *(const B128*)(bp + 16);
            acc[nt] = __builtin_amdgcn_wmma_f32_16x16x32_f16(
                false, A.v, false, Bf.v, (short)0, acc[nt], false, false);
        }
    }
#pragma unroll
    for (int nt = 0; nt < 4; nt++) {
        int n = n0 + nt * 16 + (lane & 15);
        int head = n >> 5, dd = n & 31;
        float bn = bq[n];
#pragma unroll
        for (int r = 0; r < 8; r++) {
            int grow = rowBase + r + ((lane >> 4) << 3);
            int win = grow / 49, t = grow - win * 49;
            float v = (acc[nt][r] + bn) * 0.17677669529663689f; // d^-0.5
            Q[(((long long)win * 8 + head) * 64 + t) * 32 + dd] = (_Float16)v;
        }
    }
}

// ---------- KV projection: [K | V] = gather(query) @ Wkv^T ----------
__global__ __launch_bounds__(128)
void kvproj_kernel(const float* __restrict__ X, const _Float16* __restrict__ Wt,
                   const float* __restrict__ bkv, _Float16* __restrict__ K,
                   _Float16* __restrict__ Vt) {
    const int tid = threadIdx.x, lane = tid & 31, wave = tid >> 5;
    const int rowBase = blockIdx.x * 16;
    __shared__ _Float16 As[16][32];

    const int lr = tid >> 3;
    const int lk = (tid & 7) * 4;
    const float* aRow = X + gatherRow(rowBase + lr) + lk;

    v8f acc[8];
    for (int i = 0; i < 8; i++) for (int j = 0; j < 8; j++) acc[i][j] = 0.f;

    const int arow = lane & 15;
    const int kbA  = (lane >> 4) * 8;
    const int kbB  = (lane >> 4) * 8;
    const int n0   = wave * 128;

    for (int kk = 0; kk < 256; kk += 32) {
        __syncthreads();
        float4 f = *(const float4*)(aRow + kk);
        if (kk + 32 < 256) __builtin_prefetch(aRow + kk + 32, 0, 3);
        As[lr][lk + 0] = (_Float16)f.x;
        As[lr][lk + 1] = (_Float16)f.y;
        As[lr][lk + 2] = (_Float16)f.z;
        As[lr][lk + 3] = (_Float16)f.w;
        __syncthreads();
        Frag A;
        A.b[0] = *(const B128*)&As[arow][kbA];
        A.b[1] = *(const B128*)&As[arow][kbA + 16];
#pragma unroll
        for (int nt = 0; nt < 8; nt++) {
            int n = n0 + nt * 16 + (lane & 15);
            const _Float16* bp = Wt + n * 256 + kk + kbB;
            Frag Bf;
            Bf.b[0] = *(const B128*)bp;
            Bf.b[1] = *(const B128*)(bp + 16);
            acc[nt] = __builtin_amdgcn_wmma_f32_16x16x32_f16(
                false, A.v, false, Bf.v, (short)0, acc[nt], false, false);
        }
    }
#pragma unroll
    for (int nt = 0; nt < 8; nt++) {
        int n = n0 + nt * 16 + (lane & 15);
        float bn = bkv[n];
#pragma unroll
        for (int r = 0; r < 8; r++) {
            int grow = rowBase + r + ((lane >> 4) << 3);
            int win = grow / 49, t = grow - win * 49;
            float v = acc[nt][r] + bn;
            if (n < 256) {                      // K: [win][h][t][d]
                int head = n >> 5, dd = n & 31;
                K[(((long long)win * 8 + head) * 64 + t) * 32 + dd] = (_Float16)v;
            } else {                            // V transposed: [win][h][d][t]
                int nv = n - 256, head = nv >> 5, dd = nv & 31;
                Vt[(((long long)win * 8 + head) * 32 + dd) * 64 + t] = (_Float16)v;
            }
        }
    }
}

// ---------- attention: S = qk^T + bias + mask, softmax, O = P v ----------
__global__ __launch_bounds__(128)
void attn_kernel(const _Float16* __restrict__ Q, const _Float16* __restrict__ K,
                 const _Float16* __restrict__ Vt, const float* __restrict__ biasArr,
                 _Float16* __restrict__ O) {
    const int win = blockIdx.x, h = blockIdx.y;
    const int tid = threadIdx.x, lane = tid & 31, wave = tid >> 5;
    const long long base = ((long long)win * 8 + h) * 2048;  // 64*32 elements
    const _Float16* Qb = Q + base;
    const _Float16* Kb = K + base;
    const _Float16* Vb = Vt + base;                           // [32][64]

    __shared__ _Float16 Ks[64 * 32];          // 4KB staged K tile
    __shared__ _Float16 Vs[32 * 64];          // 4KB staged V^T tile
    __shared__ _Float16 Pl[4][16][64];

    // --- async stage K and V into LDS (ASYNCcnt path), 16B per lane per op ---
    {
        unsigned ldsK = (unsigned)(uintptr_t)(void*)Ks;  // low 32b = LDS offset
        unsigned ldsV = (unsigned)(uintptr_t)(void*)Vs;
        const char* gk = (const char*)Kb;
        const char* gv = (const char*)Vb;
#pragma unroll
        for (int j = 0; j < 2; ++j) {
            int c = (tid + j * 128) * 16;     // byte offset, covers 4KB
            unsigned long long gka = (unsigned long long)(uintptr_t)(gk + c);
            unsigned lka = ldsK + c;
            asm volatile("global_load_async_to_lds_b128 %0, %1, off"
                         :: "v"(lka), "v"(gka) : "memory");
            unsigned long long gva = (unsigned long long)(uintptr_t)(gv + c);
            unsigned lva = ldsV + c;
            asm volatile("global_load_async_to_lds_b128 %0, %1, off"
                         :: "v"(lva), "v"(gva) : "memory");
        }
        asm volatile("s_wait_asynccnt 0x0" ::: "memory");
    }
    __syncthreads();

    const int kbA = (lane >> 4) * 8;
    const int kbB = (lane >> 4) * 8;
    const int ncol = lane & 15;

    // S tiles: this wave owns rows [wave*16, wave*16+16)
    Frag Aq;
    {
        const _Float16* qp = Qb + (wave * 16 + (lane & 15)) * 32;
        Aq.b[0] = *(const B128*)(qp + kbA);
        Aq.b[1] = *(const B128*)(qp + kbA + 16);
    }
    v8f s[4];
#pragma unroll
    for (int nt = 0; nt < 4; nt++) {
        int n = nt * 16 + ncol;
        const _Float16* kp = Ks + n * 32 + kbB;
        Frag Bk;
        Bk.b[0] = *(const B128*)kp;
        Bk.b[1] = *(const B128*)(kp + 16);
        v8f z; for (int j = 0; j < 8; j++) z[j] = 0.f;
        s[nt] = __builtin_amdgcn_wmma_f32_16x16x32_f16(
            false, Aq.v, false, Bk.v, (short)0, z, false, false);
    }

    // bias + shift mask (region ids computed from window coords)
    int wi = win & 255, wy = wi >> 4, wx = wi & 15;
#pragma unroll
    for (int nt = 0; nt < 4; nt++) {
        int n = nt * 16 + ncol;
#pragma unroll
        for (int r = 0; r < 8; r++) {
            if (n >= 49) { s[nt][r] = -1e30f; continue; }
            int m = wave * 16 + r + ((lane >> 4) << 3);
            float v = s[nt][r] + biasArr[(h << 12) + ((m & 63) << 6) + n];
            int ym = m / 7, xm = m - ym * 7;
            int pym = wy * 7 + ym, pxm = wx * 7 + xm;
            int idm = ((pym < 105) ? 0 : ((pym < 109) ? 1 : 2)) * 3
                    + ((pxm < 105) ? 0 : ((pxm < 109) ? 1 : 2));
            int yn = n / 7, xn = n - yn * 7;
            int pyn = wy * 7 + yn, pxn = wx * 7 + xn;
            int idn = ((pyn < 105) ? 0 : ((pyn < 109) ? 1 : 2)) * 3
                    + ((pxn < 105) ? 0 : ((pxn < 109) ? 1 : 2));
            if (idm != idn) v -= 100.f;
            s[nt][r] = v;
        }
    }

    // row softmax: a row spans 16 lanes within one half-wave (C-layout)
    float rs[8];
#pragma unroll
    for (int r = 0; r < 8; r++) {
        float mx = fmaxf(fmaxf(s[0][r], s[1][r]), fmaxf(s[2][r], s[3][r]));
        for (int off = 8; off > 0; off >>= 1)
            mx = fmaxf(mx, __shfl_xor(mx, off, 16));
        float sum = 0.f;
#pragma unroll
        for (int nt = 0; nt < 4; nt++) {
            float p = __expf(s[nt][r] - mx);
            s[nt][r] = p;
            sum += p;
        }
        for (int off = 8; off > 0; off >>= 1)
            sum += __shfl_xor(sum, off, 16);
        rs[r] = sum;
    }

    // re-layout P (C-layout regs) -> LDS row-major for A-fragments (wave-private)
#pragma unroll
    for (int nt = 0; nt < 4; nt++) {
        int col = nt * 16 + ncol;
#pragma unroll
        for (int r = 0; r < 8; r++) {
            int rloc = r + ((lane >> 4) << 3);
            Pl[wave][rloc][col] = (_Float16)s[nt][r];
        }
    }

    // O = P @ V  (K=64 in two steps; B-fragments from staged V^T are contiguous)
    v8f o[2];
    for (int i = 0; i < 2; i++) for (int j = 0; j < 8; j++) o[i][j] = 0.f;
#pragma unroll
    for (int kc = 0; kc < 64; kc += 32) {
        Frag Ap;
        Ap.b[0] = *(const B128*)&Pl[wave][lane & 15][kc + kbA];
        Ap.b[1] = *(const B128*)&Pl[wave][lane & 15][kc + kbA + 16];
#pragma unroll
        for (int nt = 0; nt < 2; nt++) {
            int n = nt * 16 + ncol;                 // d index
            const _Float16* vp = Vs + n * 64 + kc + kbB;
            Frag Bv;
            Bv.b[0] = *(const B128*)vp;
            Bv.b[1] = *(const B128*)(vp + 16);
            o[nt] = __builtin_amdgcn_wmma_f32_16x16x32_f16(
                false, Ap.v, false, Bv.v, (short)0, o[nt], false, false);
        }
    }
#pragma unroll
    for (int nt = 0; nt < 2; nt++) {
        int dd = nt * 16 + ncol;
#pragma unroll
        for (int r = 0; r < 8; r++) {
            int m = wave * 16 + r + ((lane >> 4) << 3);
            if (m < 49)
                O[((long long)win * 64 + m) * 256 + h * 32 + dd] =
                    (_Float16)(o[nt][r] / rs[r]);
        }
    }
}

// ---------- output projection + inverse roll scatter ----------
__global__ __launch_bounds__(128)
void proj_kernel(const _Float16* __restrict__ O, const _Float16* __restrict__ Wt,
                 const float* __restrict__ bp, float* __restrict__ out) {
    const int tid = threadIdx.x, lane = tid & 31, wave = tid >> 5;
    const int rowBase = blockIdx.x * 16;
    const int kbA = (lane >> 4) * 8;
    const int kbB = (lane >> 4) * 8;
    const int n0 = wave * 64;

    int arowg = rowBase + (lane & 15);
    int winA = arowg / 49, tA = arowg - winA * 49;
    const _Float16* ap = O + ((long long)winA * 64 + tA) * 256;

    v8f acc[4];
    for (int i = 0; i < 4; i++) for (int j = 0; j < 8; j++) acc[i][j] = 0.f;

    for (int kk = 0; kk < 256; kk += 32) {
        Frag A;
        A.b[0] = *(const B128*)(ap + kk + kbA);
        A.b[1] = *(const B128*)(ap + kk + kbA + 16);
        if (kk + 32 < 256) __builtin_prefetch(ap + kk + 32 + kbA, 0, 3);
#pragma unroll
        for (int nt = 0; nt < 4; nt++) {
            int n = n0 + nt * 16 + (lane & 15);
            const _Float16* bpp = Wt + n * 256 + kk + kbB;
            Frag Bf;
            Bf.b[0] = *(const B128*)bpp;
            Bf.b[1] = *(const B128*)(bpp + 16);
            acc[nt] = __builtin_amdgcn_wmma_f32_16x16x32_f16(
                false, A.v, false, Bf.v, (short)0, acc[nt], false, false);
        }
    }
#pragma unroll
    for (int nt = 0; nt < 4; nt++) {
        int n = n0 + nt * 16 + (lane & 15);
        float bn = bp[n];
#pragma unroll
        for (int r = 0; r < 8; r++) {
            int grow = rowBase + r + ((lane >> 4) << 3);
            int win = grow / 49, t = grow - win * 49;
            int b = win >> 8, wi = win & 255;
            int wy = wi >> 4, wx = wi & 15;
            int ty = t / 7, tx = t - ty * 7;
            int dy = wy * 7 + ty + 3; if (dy >= 112) dy -= 112;  // roll(+3)
            int dx = wx * 7 + tx + 3; if (dx >= 112) dx -= 112;
            out[((long long)b * 12544 + dy * 112 + dx) * 256 + n] = acc[nt][r] + bn;
        }
    }
}

extern "C" void kernel_launch(void* const* d_in, const int* in_sizes, int n_in,
                              void* d_out, int out_size, void* d_ws, size_t ws_size,
                              hipStream_t stream) {
    const float* query   = (const float*)d_in[0];   // -> K, V
    const float* query2  = (const float*)d_in[1];   // -> Q
    const float* Wq      = (const float*)d_in[2];
    const float* bq      = (const float*)d_in[3];
    const float* Wkv     = (const float*)d_in[4];
    const float* bkv     = (const float*)d_in[5];
    const float* Wproj   = (const float*)d_in[6];
    const float* bproj   = (const float*)d_in[7];
    const float* btab    = (const float*)d_in[8];
    float* out = (float*)d_out;

    char* ws = (char*)d_ws;
    _Float16* Wqt   = (_Float16*)(ws + OFF_WQT);
    _Float16* Wkvt  = (_Float16*)(ws + OFF_WKVT);
    _Float16* Wpt   = (_Float16*)(ws + OFF_WPT);
    float*    biasA = (float*)   (ws + OFF_BIAS);
    _Float16* Qb    = (_Float16*)(ws + OFF_Q);
    _Float16* Kb    = (_Float16*)(ws + OFF_K);
    _Float16* Vtb   = (_Float16*)(ws + OFF_VT);
    _Float16* Ob    = (_Float16*)(ws + OFF_O);

    prep_kernel<<<512, 256, 0, stream>>>(Wq, Wkv, Wproj, btab, Wqt, Wkvt, Wpt, biasA);
    qproj_kernel<<<NBLK, 128, 0, stream>>>(query2, Wqt, bq, Qb);
    kvproj_kernel<<<NBLK, 128, 0, stream>>>(query, Wkvt, bkv, Kb, Vtb);
    attn_kernel<<<dim3(2048, 8), 128, 0, stream>>>(Qb, Kb, Vtb, biasA, Ob);
    proj_kernel<<<NBLK, 128, 0, stream>>>(Ob, Wpt, bproj, out);
}
